// MANNNTM_82265803587621
// MI455X (gfx1250) — compile-verified, matrix operand order checked by reading, not run on hardware
//
#include <hip/hip_runtime.h>
#include <hip/hip_bf16.h>

typedef _Float16 v16h __attribute__((ext_vector_type(16)));
typedef _Float16 v8h  __attribute__((ext_vector_type(8)));
typedef float    v8f  __attribute__((ext_vector_type(8)));
typedef float    v4f  __attribute__((ext_vector_type(4)));

#define B_    256
#define T_    100
#define IN_   784
#define H_    200
#define G4_   800
#define R_    4
#define N_    128
#define D_    40
#define NC_   5
#define GAMMA_ 0.95f

// ---------------- f32 -> f16 weight conversion with K padding ----------------
__global__ void k_cvt_w(const float* __restrict__ src, _Float16* __restrict__ dst,
                        int rows, int cols, int kpad) {
  int idx = blockIdx.x * blockDim.x + threadIdx.x;
  if (idx >= rows * kpad) return;
  int r = idx / kpad, k = idx % kpad;
  dst[idx] = (k < cols) ? (_Float16)src[(size_t)r * cols + k] : (_Float16)0.f;
}

__global__ void k_zero(float* __restrict__ p, int n) {
  int idx = blockIdx.x * blockDim.x + threadIdx.x;
  if (idx < n) p[idx] = 0.f;
}

// ---------------- WMMA fragment helpers ----------------
// Unconditional: 8 contiguous f32 -> 8 f16.
__device__ inline v8h cvt8(const float* p) {
  v4f a = *(const v4f*)p;
  v4f b = *(const v4f*)(p + 4);
  v8h r;
  r[0] = (_Float16)a[0]; r[1] = (_Float16)a[1];
  r[2] = (_Float16)a[2]; r[3] = (_Float16)a[3];
  r[4] = (_Float16)b[0]; r[5] = (_Float16)b[1];
  r[6] = (_Float16)b[2]; r[7] = (_Float16)b[3];
  return r;
}
// Guarded (tail chunk only): zero when the 8-group lies beyond valid K.
__device__ inline v8h cvt8g(const float* p, bool valid) {
  v8h r = {0, 0, 0, 0, 0, 0, 0, 0};
  if (valid) r = cvt8(p);
  return r;
}

__device__ inline v16h cat16(v8h lo, v8h hi) {
  return __builtin_shufflevector(lo, hi, 0, 1, 2, 3, 4, 5, 6, 7,
                                 8, 9, 10, 11, 12, 13, 14, 15);
}

__device__ inline v8f wmma_f16(v16h a, v16h b, v8f c) {
  return __builtin_amdgcn_wmma_f32_16x16x32_f16(
      /*neg_a=*/false, a, /*neg_b=*/false, b,
      /*c_mod=*/(short)0, c, /*reuse_a=*/false, /*reuse_b=*/false);
}

// ---------------- Generic WMMA GEMM, NT column tiles per wave ----------------
// C[i,j] = (Cin ? Cin[i*ldci+j] : 0) + bias[j] + sum_k A[i,k] * Bh[j,k]
// A: f32, row stride lda, rows = gridDim.y*16 (all valid), converted to f16 on load.
// Bh: f16 row-major (ncols x kpad), kpad % 32 == 0, zero-padded for k >= kvalid.
// kvalid % 8 == 0 so every 8-element A group is all-valid or all-invalid.
// One wave computes NT adjacent 16x16 output tiles, sharing the A fragment
// (A-operand reuse = NT -> NT-fold cut in f32 activation HBM traffic).
// Main K loop is branch-free (exec stays all-1s through v_wmma); only the single
// tail chunk carries A-validity guards. Out-of-range column tiles clamp their B
// row pointer (loads discarded at the store guard) to avoid exec-mask churn.
template <int NT>
__global__ __launch_bounds__(32)
void k_gemm(const float* __restrict__ A, int lda,
            const _Float16* __restrict__ Bh, int kpad, int kvalid,
            const float* __restrict__ Cin, int ldci,
            const float* __restrict__ bias,
            float* __restrict__ C, int ldc, int ncols) {
  const int lane = threadIdx.x & 31;
  const int hs   = lane >> 4;   // lane half select (K sub-block)
  const int l15  = lane & 15;
  const int tm   = blockIdx.y;  // row tile

  const int rowA = tm * 16 + l15;
  const float* arow = A + (size_t)rowA * lda;

  const _Float16* brow[NT];
  v8f acc[NT];
  const v8f vzero = {0.f, 0.f, 0.f, 0.f, 0.f, 0.f, 0.f, 0.f};
#pragma unroll
  for (int t = 0; t < NT; t++) {
    int colB = (blockIdx.x * NT + t) * 16 + l15;
    int cb = (colB < ncols) ? colB : (ncols - 1);  // clamp: loads, not stores
    brow[t] = Bh + (size_t)cb * kpad;
    acc[t] = vzero;
  }

  int k0 = 0;
  // ---- branch-free main loop: all A groups valid ----
  for (; k0 + 32 <= kvalid; k0 += 32) {
    const int g0 = k0 + hs * 8;        // fragment elements 0..7
    const int g1 = k0 + 16 + hs * 8;   // fragment elements 8..15
    v16h av = cat16(cvt8(arow + g0), cvt8(arow + g1));
#pragma unroll
    for (int t = 0; t < NT; t++) {
      v16h bv = cat16(*(const v8h*)(brow[t] + g0),
                      *(const v8h*)(brow[t] + g1));
      acc[t] = wmma_f16(av, bv, acc[t]);
    }
  }
  // ---- tail chunk(s): A groups guarded, B already zero-padded in memory ----
  for (; k0 < kpad; k0 += 32) {
    const int g0 = k0 + hs * 8;
    const int g1 = k0 + 16 + hs * 8;
    v16h av = cat16(cvt8g(arow + g0, g0 + 8 <= kvalid),
                    cvt8g(arow + g1, g1 + 8 <= kvalid));
#pragma unroll
    for (int t = 0; t < NT; t++) {
      v16h bv = cat16(*(const v8h*)(brow[t] + g0),
                      *(const v8h*)(brow[t] + g1));
      acc[t] = wmma_f16(av, bv, acc[t]);
    }
  }

  // ---- epilogue ----
#pragma unroll
  for (int t = 0; t < NT; t++) {
    const int colB = (blockIdx.x * NT + t) * 16 + l15;
    if (colB < ncols) {
      const float bj = bias ? bias[colB] : 0.f;
#pragma unroll
      for (int e = 0; e < 8; e++) {
        const int i = tm * 16 + hs * 8 + e;  // C/D layout: VGPR e -> M = e + 8*hs
        float v = acc[t][e] + bj;
        if (Cin) v += Cin[(size_t)i * ldci + colB];
        C[(size_t)i * ldc + colB] = v;
      }
    }
  }
}

// ---------------- LSTM elementwise ----------------
__global__ void k_lstm(const float* __restrict__ gates,
                       float* __restrict__ c, float* __restrict__ h) {
  int idx = blockIdx.x * blockDim.x + threadIdx.x;
  if (idx >= B_ * H_) return;
  int b = idx / H_, j = idx % H_;
  const float* g = gates + (size_t)b * G4_;
  float ig = 1.f / (1.f + expf(-g[j]));
  float fg = 1.f / (1.f + expf(-g[H_ + j]));
  float gg = tanhf(g[2 * H_ + j]);
  float og = 1.f / (1.f + expf(-g[3 * H_ + j]));
  float ct = fg * c[idx] + ig * gg;
  c[idx] = ct;
  h[idx] = og * tanhf(ct);
}

// ---------------- NTM memory module: one block (128 thr) per batch element ----
__global__ __launch_bounds__(128)
void k_mem(int t,
           const float* __restrict__ key, const float* __restrict__ add,
           const float* __restrict__ h,
           float* __restrict__ wr, float* __restrict__ wu, float* __restrict__ M,
           float* __restrict__ rbuf,
           const float* __restrict__ Wsig, const float* __restrict__ bsig,
           const float* __restrict__ Who,  const float* __restrict__ bho,
           const float* __restrict__ Wro,  const float* __restrict__ bro,
           float* __restrict__ out) {
  const int b = blockIdx.x;
  const int tid = threadIdx.x;

  __shared__ float s_key[R_ * D_];
  __shared__ float s_add[R_ * D_];
  __shared__ float s_h[H_];
  __shared__ float s_Mt[N_][D_];     // 20 KB
  __shared__ float s_K[R_][N_];
  __shared__ float s_wrt[R_][N_];
  __shared__ float s_rt[R_ * D_];
  __shared__ float s_red[R_][2];     // softmax {max, sum}
  __shared__ float s_scr[128];
  __shared__ float s_x2[R_];
  __shared__ float s_sigma;
  __shared__ int   s_lu[R_];

  for (int i = tid; i < R_ * D_; i += 128) {
    s_key[i] = key[(size_t)b * R_ * D_ + i];
    s_add[i] = add[(size_t)b * R_ * D_ + i];
  }
  for (int i = tid; i < H_; i += 128) s_h[i] = h[(size_t)b * H_ + i];
  __syncthreads();

  // sigma = h . Wsig + bsig  (raw linear gate, as in the reference)
  float p = 0.f;
  for (int j = tid; j < H_; j += 128) p += s_h[j] * Wsig[j];
  s_scr[tid] = p;
  __syncthreads();
  for (int s = 64; s > 0; s >>= 1) {
    if (tid < s) s_scr[tid] += s_scr[tid + s];
    __syncthreads();
  }
  if (tid == 0) s_sigma = s_scr[0] + bsig[0];
  if (tid < R_) {  // key row squared norms
    float s = 0.f;
    for (int d = 0; d < D_; d++) { float v = s_key[tid * D_ + d]; s += v * v; }
    s_x2[tid] = s;
  }
  if (tid == 0) {
    // stable argmin x4 == first R of stable ascending argsort of wu
    for (int r = 0; r < R_; r++) {
      float best = INFINITY; int bi = 0;
      for (int n = 0; n < N_; n++) {
        bool used = false;
        for (int q = 0; q < r; q++) if (s_lu[q] == n) used = true;
        float v = wu[(size_t)b * N_ + n];
        if (!used && v < best) { best = v; bi = n; }
      }
      s_lu[r] = bi;
    }
  }
  __syncthreads();

  const int n = tid;  // one thread per memory slot
  const float sigma = s_sigma;
  const float wlu_n =
      (n == s_lu[0] || n == s_lu[1] || n == s_lu[2] || n == s_lu[3]) ? 1.f : 0.f;

  float ww[R_];
  float wwsum = 0.f;
#pragma unroll
  for (int r = 0; r < R_; r++) {
    float wrold = wr[((size_t)b * R_ + r) * N_ + n];
    ww[r] = sigma * wrold + (1.f - sigma) * wlu_n;
    wwsum += ww[r];
  }

  // M_t row n, squared norm, cosine numerators
  float y2 = 0.f;
  float* Mrow = M + ((size_t)b * N_ + n) * D_;
  float Kdot[R_] = {0.f, 0.f, 0.f, 0.f};
#pragma unroll 8
  for (int d = 0; d < D_; d++) {
    float m = Mrow[d] * wlu_n;
#pragma unroll
    for (int r = 0; r < R_; r++) m += ww[r] * s_add[r * D_ + d];
    s_Mt[n][d] = m;
    Mrow[d] = m;
    y2 += m * m;
#pragma unroll
    for (int r = 0; r < R_; r++) Kdot[r] += s_key[r * D_ + d] * m;
  }
#pragma unroll
  for (int r = 0; r < R_; r++)
    s_K[r][n] = Kdot[r] / sqrtf(s_x2[r] * y2 + 1e-6f);
  __syncthreads();

  // softmax over N per read head: wave w reduces row r=w
  {
    const int w = tid >> 5, lane = tid & 31;
    float mx = -INFINITY;
    for (int q = lane; q < N_; q += 32) mx = fmaxf(mx, s_K[w][q]);
    for (int o = 16; o > 0; o >>= 1) mx = fmaxf(mx, __shfl_xor(mx, o, 32));
    float sm = 0.f;
    for (int q = lane; q < N_; q += 32) sm += expf(s_K[w][q] - mx);
    for (int o = 16; o > 0; o >>= 1) sm += __shfl_xor(sm, o, 32);
    if (lane == 0) { s_red[w][0] = mx; s_red[w][1] = sm; }
  }
  __syncthreads();

  float wun = wu[(size_t)b * N_ + n];
  float wrtsum = 0.f;
#pragma unroll
  for (int r = 0; r < R_; r++) {
    float v = expf(s_K[r][n] - s_red[r][0]) / s_red[r][1];
    s_wrt[r][n] = v;
    wr[((size_t)b * R_ + r) * N_ + n] = v;
    wrtsum += v;
  }
  wu[(size_t)b * N_ + n] = GAMMA_ * wun + wrtsum + wwsum;
  __syncthreads();

  // r_t[r,d] = sum_n wr_t[r,n] * M_t[n,d]
  for (int idx = tid; idx < R_ * D_; idx += 128) {
    int r = idx / D_, d = idx % D_;
    float s = 0.f;
    for (int q = 0; q < N_; q++) s += s_wrt[r][q] * s_Mt[q][d];
    s_rt[idx] = s;
    rbuf[(size_t)b * R_ * D_ + idx] = s;
  }
  __syncthreads();

  // out[b,t,:] = h.Who^T + bho + r.Wro^T + bro
  if (tid < NC_) {
    float o = bho[tid] + bro[tid];
    for (int j = 0; j < H_; j++)      o += s_h[j]  * Who[tid * H_ + j];
    for (int k = 0; k < R_ * D_; k++) o += s_rt[k] * Wro[tid * R_ * D_ + k];
    out[((size_t)b * T_ + t) * NC_ + tid] = o;
  }
}

// ---------------- host ----------------
extern "C" void kernel_launch(void* const* d_in, const int* in_sizes, int n_in,
                              void* d_out, int out_size, void* d_ws, size_t ws_size,
                              hipStream_t stream) {
  (void)in_sizes; (void)n_in; (void)out_size; (void)ws_size;
  const float* x    = (const float*)d_in[0];
  const float* Wkey = (const float*)d_in[1];
  const float* bkey = (const float*)d_in[2];
  const float* Wadd = (const float*)d_in[3];
  const float* badd = (const float*)d_in[4];
  const float* Wsig = (const float*)d_in[5];
  const float* bsig = (const float*)d_in[6];
  const float* Who  = (const float*)d_in[7];
  const float* bho  = (const float*)d_in[8];
  const float* Wro  = (const float*)d_in[9];
  const float* bro  = (const float*)d_in[10];
  const float* Wrh  = (const float*)d_in[11];
  const float* brh  = (const float*)d_in[12];
  const float* Wih  = (const float*)d_in[13];
  const float* bih  = (const float*)d_in[14];
  const float* Whh  = (const float*)d_in[15];
  const float* bhh  = (const float*)d_in[16];
  float* out = (float*)d_out;

  char* ws = (char*)d_ws;
  size_t o = 0;
  auto take = [&](size_t bytes) -> char* {
    char* p = ws + o;
    o += (bytes + 255) & ~(size_t)255;
    return p;
  };
  _Float16* Wih_h  = (_Float16*)take((size_t)800 * 800 * 2);  // K pad 784->800
  _Float16* Whh_h  = (_Float16*)take((size_t)800 * 224 * 2);  // K pad 200->224
  _Float16* Wrh_h  = (_Float16*)take((size_t)200 * 160 * 2);  // K = 160
  _Float16* Wkey_h = (_Float16*)take((size_t)160 * 224 * 2);
  _Float16* Wadd_h = (_Float16*)take((size_t)160 * 224 * 2);
  float* G     = (float*)take((size_t)B_ * T_ * G4_ * 4);     // precomputed x@Wih^T+bih
  float* gates = (float*)take((size_t)B_ * G4_ * 4);
  float* keyb  = (float*)take((size_t)B_ * R_ * D_ * 4);
  float* addb  = (float*)take((size_t)B_ * R_ * D_ * 4);
  const int stateFloats = B_ * H_ + B_ * H_ + B_ * R_ * D_ +
                          B_ * R_ * N_ + B_ * N_ + B_ * N_ * D_;
  float* state = (float*)take((size_t)stateFloats * 4);
  float* hS  = state;
  float* cS  = hS  + B_ * H_;
  float* rS  = cS  + B_ * H_;
  float* wrS = rS  + B_ * R_ * D_;
  float* wuS = wrS + B_ * R_ * N_;
  float* MS  = wuS + B_ * N_;

  // weight conversion + state zeroing (every call -> deterministic)
  k_cvt_w<<<(800 * 800 + 255) / 256, 256, 0, stream>>>(Wih,  Wih_h,  800, IN_, 800);
  k_cvt_w<<<(800 * 224 + 255) / 256, 256, 0, stream>>>(Whh,  Whh_h,  800, H_,  224);
  k_cvt_w<<<(200 * 160 + 255) / 256, 256, 0, stream>>>(Wrh,  Wrh_h,  200, 160, 160);
  k_cvt_w<<<(160 * 224 + 255) / 256, 256, 0, stream>>>(Wkey, Wkey_h, 160, H_,  224);
  k_cvt_w<<<(160 * 224 + 255) / 256, 256, 0, stream>>>(Wadd, Wadd_h, 160, H_,  224);
  k_zero<<<(stateFloats + 255) / 256, 256, 0, stream>>>(state, stateFloats);

  // Phase 1: G[b*T+t, :] = x[b,t,:] @ Wih^T + bih   (25600 x 784 x 800)
  // 50 column tiles / NT=4 per wave -> grid.x = 13
  k_gemm<4><<<dim3(13, (B_ * T_) / 16), 32, 0, stream>>>(
      x, IN_, Wih_h, 800, IN_, nullptr, 0, bih, G, G4_, G4_);

  // Phase 2: recurrence (batch rows tiled 16-wide for WMMA)
  for (int t = 0; t < T_; t++) {
    // h = h + r @ Wrh^T + brh          (13 tiles -> 4 groups)
    k_gemm<4><<<dim3(4, B_ / 16), 32, 0, stream>>>(
        rS, R_ * D_, Wrh_h, 160, 160, hS, H_, brh, hS, H_, H_);
    // gates = G[:,t,:] + h @ Whh^T + bhh  (50 tiles -> 13 groups)
    k_gemm<4><<<dim3(13, B_ / 16), 32, 0, stream>>>(
        hS, H_, Whh_h, 224, H_, G + (size_t)t * G4_, T_ * G4_, bhh, gates, G4_, G4_);
    k_lstm<<<(B_ * H_ + 255) / 256, 256, 0, stream>>>(gates, cS, hS);
    // key / add projections               (10 tiles -> 3 groups)
    k_gemm<4><<<dim3(3, B_ / 16), 32, 0, stream>>>(
        hS, H_, Wkey_h, 224, H_, nullptr, 0, bkey, keyb, R_ * D_, R_ * D_);
    k_gemm<4><<<dim3(3, B_ / 16), 32, 0, stream>>>(
        hS, H_, Wadd_h, 224, H_, nullptr, 0, badd, addb, R_ * D_, R_ * D_);
    // memory module + output head
    k_mem<<<B_, 128, 0, stream>>>(t, keyb, addb, hS, wrS, wuS, MS, rS,
                                  Wsig, bsig, Who, bho, Wro, bro, out);
  }
}
